// ARMA_49289044689240
// MI455X (gfx1250) — compile-verified, matrix operand order checked by reading, not run on hardware
//
#include <hip/hip_runtime.h>
#include <hip/hip_bf16.h>

// ---------------------------------------------------------------------------
// ARMA GNN for MI455X (gfx1250): bf16 WMMA GEMMs + fp32 scatter propagate.
// ---------------------------------------------------------------------------

typedef __attribute__((ext_vector_type(16))) __bf16 v16bf;
typedef __attribute__((ext_vector_type(8)))  float  v8f;

#define NNODES 30000
#define NEDGES 300000
#define FIN    256
#define FHID   256
#define FOUT   64
#define KST    3

static inline long cdivl(long a, long b) { return (a + b - 1) / b; }

// ---------------------------------------------------------------------------
// Elementwise / graph kernels
// ---------------------------------------------------------------------------

__global__ void zero_f32_kernel(float* p, long n) {
    long i = (long)blockIdx.x * blockDim.x + threadIdx.x;
    if (i < n) p[i] = 0.0f;
}

__global__ void deg_kernel(const int* ei, float* deg, int E) {
    int e = blockIdx.x * blockDim.x + threadIdx.x;
    if (e >= E) return;
    int c = ei[E + e];  // col = edge_index[1]
    unsafeAtomicAdd(&deg[c], 1.0f);
}

__global__ void norm_kernel(const int* ei, const float* deg, float* nrm, int E) {
    int e = blockIdx.x * blockDim.x + threadIdx.x;
    if (e >= E) return;
    int r = ei[e];
    int c = ei[E + e];
    float dr = deg[r], dc = deg[c];
    float ir = (dr > 0.0f) ? rsqrtf(fmaxf(dr, 1.0f)) : 0.0f;
    float ic = (dc > 0.0f) ? rsqrtf(fmaxf(dc, 1.0f)) : 0.0f;
    nrm[e] = ir * ic;
}

// out[k, col[e], :] += norm[e] * in[k, row[e], :], float4 per thread
__global__ void propagate_kernel(const float* __restrict__ hin, float* __restrict__ hout,
                                 const int* __restrict__ ei, const float* __restrict__ nrm,
                                 int n, int E, int f, int c4shift) {
    long tid = (long)blockIdx.x * blockDim.x + threadIdx.x;
    int c4 = 1 << c4shift;
    long total = (long)E * c4;
    if (tid >= total) return;
    int k = blockIdx.z;
    int e  = (int)(tid >> c4shift);
    int fc = (int)(tid & (c4 - 1));
    float nv = nrm[e];
    int r = ei[e];
    int c = ei[E + e];
    const float4 v = *(const float4*)(hin + ((long)k * n + r) * f + fc * 4);
    float* o = hout + ((long)k * n + c) * f + fc * 4;
    unsafeAtomicAdd(o + 0, nv * v.x);
    unsafeAtomicAdd(o + 1, nv * v.y);
    unsafeAtomicAdd(o + 2, nv * v.z);
    unsafeAtomicAdd(o + 3, nv * v.w);
}

__global__ void add_act_kernel(const float* a, const float* b, float* out, long n, int do_relu) {
    long i = (long)blockIdx.x * blockDim.x + threadIdx.x;
    if (i >= n) return;
    float v = a[i] + b[i];
    out[i] = do_relu ? fmaxf(v, 0.0f) : v;
}

// out[i] = (relu?)( mean_k in[k*per + i] )
__global__ void mean_k_kernel(const float* in, float* out, long per, int nk, int do_relu) {
    long i = (long)blockIdx.x * blockDim.x + threadIdx.x;
    if (i >= per) return;
    float s = 0.0f;
    for (int k = 0; k < nk; ++k) s += in[(long)k * per + i];
    s *= (1.0f / nk);
    out[i] = do_relu ? fmaxf(s, 0.0f) : s;
}

__global__ void log_softmax_kernel(const float* in, float* out, int n, int f) {
    int r = blockIdx.x * blockDim.x + threadIdx.x;
    if (r >= n) return;
    const float* x = in + (long)r * f;
    float m = -3.402823466e38f;
    for (int j = 0; j < f; ++j) m = fmaxf(m, x[j]);
    float s = 0.0f;
    for (int j = 0; j < f; ++j) s += __expf(x[j] - m);
    float ls = __logf(s);
    float* o = out + (long)r * f;
    for (int j = 0; j < f; ++j) o[j] = x[j] - m - ls;
}

// ---------------------------------------------------------------------------
// WMMA packing.  A-tile (16x32 bf16): lane L -> half h=L>>4, row M=L&15;
// element e -> K = 16*(e>>3) + 8*h + (e&7).  B-tile (32x16): same with N=L&15.
// Packed flat layout matches tid = ((tile)*32 + lane)*16 + e exactly.
// ---------------------------------------------------------------------------

// in: [nst, n, f] f32 row-major  ->  out: [k][rt][kt][lane][e] bf16
__global__ void pack_a_kernel(const float* __restrict__ in, __bf16* __restrict__ out,
                              int nst, int n, int f) {
    long total = (long)nst * n * f;
    long tid = (long)blockIdx.x * blockDim.x + threadIdx.x;
    if (tid >= total) return;
    int e    = (int)(tid & 15);
    int lane = (int)((tid >> 4) & 31);
    long tile = tid >> 9;
    int ktper = f >> 5;
    long tpk = (long)(n >> 4) * ktper;
    int k  = (int)(tile / tpk);
    long t2 = tile % tpk;
    int rt = (int)(t2 / ktper);
    int kt = (int)(t2 % ktper);
    int hh = lane >> 4;
    int m  = rt * 16 + (lane & 15);
    int kc = kt * 32 + ((e >> 3) << 4) + hh * 8 + (e & 7);
    out[tid] = (__bf16)in[((long)k * n + m) * f + kc];
}

// w: [nk, fin, fout] f32  ->  out: [k][ot][kt][lane][e] bf16
__global__ void pack_b_kernel(const float* __restrict__ w, __bf16* __restrict__ out,
                              int nk, int fin, int fout) {
    long total = (long)nk * fin * fout;
    long tid = (long)blockIdx.x * blockDim.x + threadIdx.x;
    if (tid >= total) return;
    int e    = (int)(tid & 15);
    int lane = (int)((tid >> 4) & 31);
    long tile = tid >> 9;
    int ktper = fin >> 5;
    int otper = fout >> 4;
    long tpk = (long)ktper * otper;
    int k  = (int)(tile / tpk);
    long t2 = tile % tpk;
    int ot = (int)(t2 / ktper);
    int kt = (int)(t2 % ktper);
    int hh = lane >> 4;
    int nc = lane & 15;
    int kr = kt * 32 + ((e >> 3) << 4) + hh * 8 + (e & 7);
    out[tid] = (__bf16)w[((long)k * fin + kr) * fout + ot * 16 + nc];
}

// ---------------------------------------------------------------------------
// WMMA GEMM: C[k, n, fout] = A[k] (n x fin, packed bf16) * B[k] (packed bf16)
//            (+ bias[k, fout] if bias != nullptr)
// One wave per 16x16 output tile; 4 waves per block.
// ---------------------------------------------------------------------------
__global__ void wmma_gemm_kernel(const __bf16* __restrict__ Apk, long aStrideK,
                                 const __bf16* __restrict__ Bpk,
                                 float* __restrict__ C, const float* __restrict__ bias,
                                 int n, int fin, int fout) {
    const int lane = threadIdx.x & 31;
    const int wave = threadIdx.x >> 5;
    const int KT = fin >> 5;
    const int OT = fout >> 4;
    const int ot = blockIdx.y * (blockDim.x >> 5) + wave;
    if (ot >= OT) return;  // wave-uniform: EXEC stays all-ones for WMMA
    const int rt = blockIdx.x;
    const int k  = blockIdx.z;

    const __bf16* Ab = Apk + (long)k * aStrideK + (long)rt * KT * 512 + lane * 16;
    const __bf16* Bb = Bpk + ((long)k * OT + ot) * (long)KT * 512 + lane * 16;

    v8f acc = {};
    for (int kt = 0; kt < KT; ++kt) {
        v16bf a = *(const v16bf*)(Ab + (long)kt * 512);
        v16bf b = *(const v16bf*)(Bb + (long)kt * 512);
        acc = __builtin_amdgcn_wmma_f32_16x16x32_bf16(
            /*neg_a=*/false, a, /*neg_b=*/false, b,
            /*c_mod=*/(short)0, acc, /*reuse_a=*/false, /*reuse_b=*/false);
    }

    const int hh = lane >> 4;
    const int col = lane & 15;
    const int gc = ot * 16 + col;
    const float bv = bias ? bias[k * fout + gc] : 0.0f;
    float* Cb = C + ((long)k * n + rt * 16 + 8 * hh) * fout + gc;
#pragma unroll
    for (int r = 0; r < 8; ++r) Cb[(long)r * fout] = acc[r] + bv;
}

// ---------------------------------------------------------------------------
// Host orchestration
// ---------------------------------------------------------------------------

extern "C" void kernel_launch(void* const* d_in, const int* in_sizes, int n_in,
                              void* d_out, int out_size, void* d_ws, size_t ws_size,
                              hipStream_t stream) {
    (void)in_sizes; (void)n_in; (void)out_size; (void)ws_size;

    const float* x        = (const float*)d_in[0];
    const int*   ei       = (const int*)d_in[1];
    const float* init_w1  = (const float*)d_in[2];
    const float* w1       = (const float*)d_in[3];
    const float* root_w1  = (const float*)d_in[4];
    const float* bias1    = (const float*)d_in[5];
    const float* init_w2  = (const float*)d_in[6];
    const float* w2       = (const float*)d_in[7];
    const float* root_w2  = (const float*)d_in[8];
    const float* bias2    = (const float*)d_in[9];
    float* out = (float*)d_out;

    // ---- workspace layout (256B aligned) ----
    char* ws = (char*)d_ws;
    size_t off = 0;
    auto alloc = [&](size_t bytes) -> char* {
        char* p = ws + off;
        off += (bytes + 255) & ~(size_t)255;
        return p;
    };
    float*  deg   = (float*)alloc((size_t)NNODES * 4);
    float*  nrm   = (float*)alloc((size_t)NEDGES * 4);
    __bf16* Xpk   = (__bf16*)alloc((size_t)NNODES * FIN * 2);         // also reused for h1 pack
    __bf16* Bi1   = (__bf16*)alloc((size_t)KST * FIN * FHID * 2);     // init_w1 packed
    __bf16* Bw1   = (__bf16*)alloc((size_t)KST * FHID * FHID * 2);    // w1 packed
    __bf16* Br1   = (__bf16*)alloc((size_t)KST * FIN * FHID * 2);     // root_w1 packed
    __bf16* Bi2   = (__bf16*)alloc((size_t)KST * FHID * FOUT * 2);
    __bf16* Bw2   = (__bf16*)alloc((size_t)KST * FOUT * FOUT * 2);
    __bf16* Br2   = (__bf16*)alloc((size_t)KST * FHID * FOUT * 2);
    float*  root  = (float*)alloc((size_t)KST * NNODES * FHID * 4);   // root1, reused as root2
    float*  bufA  = (float*)alloc((size_t)KST * NNODES * FHID * 4);
    float*  bufB  = (float*)alloc((size_t)KST * NNODES * FHID * 4);
    __bf16* hPk   = (__bf16*)alloc((size_t)KST * NNODES * FHID * 2);
    float*  h1    = (float*)alloc((size_t)NNODES * FHID * 4);         // reused as pre-softmax

    const int TB = 256;
    const long bigHid = (long)KST * NNODES * FHID;
    const long bigOut = (long)KST * NNODES * FOUT;

    auto zero = [&](float* p, long cnt) {
        zero_f32_kernel<<<dim3((unsigned)cdivl(cnt, TB)), TB, 0, stream>>>(p, cnt);
    };
    auto packA = [&](const float* in, __bf16* o, int nst, int nn, int ff) {
        long tot = (long)nst * nn * ff;
        pack_a_kernel<<<dim3((unsigned)cdivl(tot, TB)), TB, 0, stream>>>(in, o, nst, nn, ff);
    };
    auto packB = [&](const float* w, __bf16* o, int fin, int fout) {
        long tot = (long)KST * fin * fout;
        pack_b_kernel<<<dim3((unsigned)cdivl(tot, TB)), TB, 0, stream>>>(w, o, KST, fin, fout);
    };
    auto gemm = [&](const __bf16* A, long aStride, const __bf16* B, float* C,
                    const float* bias, int fin, int fout) {
        int OT = fout >> 4;
        dim3 grid(NNODES / 16, (unsigned)cdivl(OT, 4), KST);
        wmma_gemm_kernel<<<grid, 128, 0, stream>>>(A, aStride, B, C, bias, NNODES, fin, fout);
    };
    auto propagate = [&](const float* hin, float* hout, int ff) {
        int c4s = (ff == 256) ? 6 : 4;  // log2(f/4)
        long tot = (long)NEDGES << c4s;
        dim3 grid((unsigned)cdivl(tot, TB), 1, KST);
        propagate_kernel<<<grid, TB, 0, stream>>>(hin, hout, ei, nrm, NNODES, NEDGES, ff, c4s);
    };
    auto addact = [&](const float* a, const float* b, float* o, long cnt, int relu) {
        add_act_kernel<<<dim3((unsigned)cdivl(cnt, TB)), TB, 0, stream>>>(a, b, o, cnt, relu);
    };

    // ---- graph norm ----
    zero(deg, NNODES);
    deg_kernel<<<dim3((unsigned)cdivl(NEDGES, TB)), TB, 0, stream>>>(ei, deg, NEDGES);
    norm_kernel<<<dim3((unsigned)cdivl(NEDGES, TB)), TB, 0, stream>>>(ei, deg, nrm, NEDGES);

    // ---- pack inputs/weights to WMMA layouts ----
    packA(x, Xpk, 1, NNODES, FIN);
    packB(init_w1, Bi1, FIN, FHID);
    packB(w1,      Bw1, FHID, FHID);
    packB(root_w1, Br1, FIN, FHID);
    packB(init_w2, Bi2, FHID, FOUT);
    packB(w2,      Bw2, FOUT, FOUT);
    packB(root_w2, Br2, FHID, FOUT);

    // ======================= ARMA conv1 (FIN -> FHID, relu) =================
    gemm(Xpk, 0, Br1, root, bias1, FIN, FHID);      // root1 = x @ root_w1 + bias1
    gemm(Xpk, 0, Bi1, bufA, nullptr, FIN, FHID);    // h = x @ init_w1
    // t = 0
    zero(bufB, bigHid);
    propagate(bufA, bufB, FHID);
    addact(bufB, root, bufA, bigHid, 1);            // h = relu(prop + root)
    // t = 1
    packA(bufA, hPk, KST, NNODES, FHID);
    gemm(hPk, (long)NNODES * FHID, Bw1, bufB, nullptr, FHID, FHID);  // h = h @ w1
    zero(bufA, bigHid);
    propagate(bufB, bufA, FHID);
    addact(bufA, root, bufB, bigHid, 1);            // h = relu(prop + root)
    // mean over stacks (+ outer relu, idempotent)
    mean_k_kernel<<<dim3((unsigned)cdivl((long)NNODES * FHID, TB)), TB, 0, stream>>>(
        bufB, h1, (long)NNODES * FHID, KST, 1);

    // ======================= ARMA conv2 (FHID -> FOUT, identity) ============
    packA(h1, Xpk, 1, NNODES, FHID);                // reuse Xpk for packed h1
    gemm(Xpk, 0, Br2, root, bias2, FHID, FOUT);     // root2 = h1 @ root_w2 + bias2
    gemm(Xpk, 0, Bi2, bufA, nullptr, FHID, FOUT);   // g = h1 @ init_w2
    // t = 0
    zero(bufB, bigOut);
    propagate(bufA, bufB, FOUT);
    addact(bufB, root, bufA, bigOut, 0);            // g = prop + root2
    // t = 1
    packA(bufA, hPk, KST, NNODES, FOUT);
    gemm(hPk, (long)NNODES * FOUT, Bw2, bufB, nullptr, FOUT, FOUT);  // g = g @ w2
    zero(bufA, bigOut);
    propagate(bufB, bufA, FOUT);
    addact(bufA, root, bufB, bigOut, 0);            // g = prop + root2
    // mean over stacks -> pre-softmax (reuse h1)
    mean_k_kernel<<<dim3((unsigned)cdivl((long)NNODES * FOUT, TB)), TB, 0, stream>>>(
        bufB, h1, (long)NNODES * FOUT, KST, 0);

    // ---- log_softmax over feature dim ----
    log_softmax_kernel<<<dim3((unsigned)cdivl(NNODES, TB)), TB, 0, stream>>>(h1, out, NNODES, FOUT);
}
